// DCN_12610023981502
// MI455X (gfx1250) — matrix operand order repaired and checked by reading, not compiled, not run on hardware
//
#include <hip/hip_runtime.h>
#include <hip/hip_bf16.h>
#include <stdint.h>

// ---------------- problem constants ----------------
static constexpr int Bz   = 16384;   // batch
static constexpr int Fz   = 26;      // fields
static constexpr int Nz   = 100000;  // rows per table
static constexpr int Mz   = 16;      // embed dim
static constexpr int NUMz = 13;      // dense features
static constexpr int Dz   = Fz * Mz + NUMz;  // 429
static constexpr int Lz   = 6;       // cross layers
static constexpr int Hz   = 1024;    // hidden
static constexpr int DPz  = 448;     // D padded to multiple of 32 (K-step of bf16 WMMA)

typedef __attribute__((ext_vector_type(16))) __bf16 v16bf;
typedef __attribute__((ext_vector_type(8)))  float  v8f;

// ---------------- kernel 1: embedding gather + concat + bf16 hi/lo split ----------------
__global__ void build_x0_kernel(const int* __restrict__ Xi, const float* __restrict__ Xv,
                                const float* __restrict__ emb,
                                float* __restrict__ x0f,
                                __bf16* __restrict__ x0hi, __bf16* __restrict__ x0lo) {
  int idx = blockIdx.x * blockDim.x + threadIdx.x;
  if (idx >= Bz * DPz) return;
  int b = idx / DPz, c = idx % DPz;
  float v = 0.f;
  if (c < NUMz) {
    v = Xv[b * NUMz + c];
  } else if (c < Dz) {
    int e = c - NUMz;
    int f = e >> 4, m = e & 15;
    long row = (long)f * Nz + Xi[b * Fz + f];   // bag size K=1 -> sum == single gather
    v = emb[row * Mz + m];
  }
  x0f[idx] = v;
  __bf16 h = (__bf16)v;
  x0hi[idx] = h;
  x0lo[idx] = (__bf16)(v - (float)h);
}

// ---------------- kernel 2: transpose + pad + bf16 hi/lo split of weights ----------------
// W: [Kvalid, Nc] row-major  ->  Wt: [Nc, Kp] row-major (zero-padded K), split hi/lo.
__global__ void cvt_wT_kernel(const float* __restrict__ W,
                              __bf16* __restrict__ hi, __bf16* __restrict__ lo,
                              int Nc, int Kp, int Kvalid) {
  int idx = blockIdx.x * blockDim.x + threadIdx.x;
  if (idx >= Nc * Kp) return;
  int n = idx / Kp, k = idx % Kp;
  float v = (k < Kvalid) ? W[(long)k * Nc + n] : 0.f;
  __bf16 h = (__bf16)v;
  hi[idx] = h;
  lo[idx] = (__bf16)(v - (float)h);
}

// ---------------- kernel 3: CrossNet (vector param), one wave32 per row ----------------
// x_{l+1} = x0 * (x_l . w_l) + b_l + x_l ; emit only xpart[b] = x_L . Wout[H:H+D]
__global__ void crossnet_kernel(const float* __restrict__ x0,
                                const float* __restrict__ cw, const float* __restrict__ cb,
                                const float* __restrict__ Wout,
                                float* __restrict__ xpart) {
  const int wid = threadIdx.x >> 5, lane = threadIdx.x & 31;
  const int row = blockIdx.x * (blockDim.x >> 5) + wid;
  if (row >= Bz) return;
  const float* xr = x0 + (long)row * DPz;
  float x0v[DPz / 32], xv[DPz / 32];
#pragma unroll
  for (int i = 0; i < DPz / 32; ++i) { x0v[i] = xr[lane + 32 * i]; xv[i] = x0v[i]; }
  for (int l = 0; l < Lz; ++l) {
    float s = 0.f;
#pragma unroll
    for (int i = 0; i < DPz / 32; ++i) {
      int c = lane + 32 * i;
      if (c < Dz) s += xv[i] * cw[l * Dz + c];
    }
#pragma unroll
    for (int off = 16; off; off >>= 1) s += __shfl_xor(s, off, 32);
#pragma unroll
    for (int i = 0; i < DPz / 32; ++i) {
      int c = lane + 32 * i;
      float bb = (c < Dz) ? cb[l * Dz + c] : 0.f;
      xv[i] = fmaf(x0v[i], s, bb + xv[i]);   // pad lanes: x0v=0, bb=0 -> stays 0
    }
  }
  float p = 0.f;
#pragma unroll
  for (int i = 0; i < DPz / 32; ++i) {
    int c = lane + 32 * i;
    if (c < Dz) p += xv[i] * Wout[Hz + c];
  }
#pragma unroll
  for (int off = 16; off; off >>= 1) p += __shfl_xor(p, off, 32);
  if (lane == 0) xpart[row] = p;
}

// ---------------- WMMA fragment loaders (CDNA5 ISA §7.12.2 layouts) ----------------
// A 16x32 bf16, lane=16*half+r holds row M=r:
//   VGPR0..3 : K = 8*half + {0..7}      (contiguous 16B)
//   VGPR4..7 : K = 16 + 8*half + {0..7} (contiguous 16B)
__device__ __forceinline__ v16bf fragA(const __bf16* __restrict__ rowp, int k, int half) {
  union { v16bf v; uint4 q[2]; } u;
  u.q[0] = *reinterpret_cast<const uint4*>(rowp + k + half * 8);
  u.q[1] = *reinterpret_cast<const uint4*>(rowp + k + 16 + half * 8);
  return u.v;
}
// B 32x16 bf16, lane=16*half+r holds column N=r:
//   VGPR0..3 : K = 8*half + {0..7} ; VGPR4..7 : K = 16 + 8*half + {0..7} (same as A)
__device__ __forceinline__ v16bf fragB(const __bf16* __restrict__ rowp, int k, int half) {
  union { v16bf v; uint4 q[2]; } u;
  u.q[0] = *reinterpret_cast<const uint4*>(rowp + k + half * 8);
  u.q[1] = *reinterpret_cast<const uint4*>(rowp + k + 16 + half * 8);
  return u.v;
}

// ---------------- kernel 4/5: bf16x3 split GEMM ----------------
// A: [Mrows, K] bf16 hi/lo ; Bt: [Nc, K] bf16 hi/lo (B stored N-major).
// Block: 256 threads = 8 waves (4 M x 2 N); block tile 128x128; wave tile 32x64 = 2x4
// WMMA tiles; 3 WMMAs/tile/K-step: Ahi*Bhi + Ahi*Blo + Alo*Bhi (f32 accumulate).
// Epilogues: (a) Chi/Clo: relu(acc+bias) re-split to bf16 hi/lo (feeds next GEMM);
//            (b) Pout: fused dot with Wv -> exclusive per-wave partial slots (deterministic).
__global__ void __launch_bounds__(256)
gemm_bf16x3_kernel(const __bf16* __restrict__ Ahi, const __bf16* __restrict__ Alo,
                   const __bf16* __restrict__ Bthi, const __bf16* __restrict__ Btlo,
                   const float* __restrict__ bias,
                   __bf16* __restrict__ Chi, __bf16* __restrict__ Clo,
                   float* __restrict__ Pout, const float* __restrict__ Wv,
                   int K, int Nc) {
  const int tid  = threadIdx.x;
  const int w    = tid >> 5, lane = tid & 31;
  const int half = lane >> 4, r = lane & 15;
  const int wm = w & 3, wn = w >> 2;
  const int row0 = blockIdx.x * 128 + wm * 32;
  const int col0 = blockIdx.y * 128 + wn * 64;

  long aoff[2], boff[4];
#pragma unroll
  for (int i = 0; i < 2; ++i) aoff[i] = (long)(row0 + 16 * i + r) * K;
#pragma unroll
  for (int j = 0; j < 4; ++j) boff[j] = (long)(col0 + 16 * j + r) * K;

  v8f acc[2][4] = {};
  for (int k = 0; k < K; k += 32) {
    if (k + 32 < K) {   // CDNA5 global_prefetch_b8 of the next K-slab
      __builtin_prefetch(Ahi + aoff[0] + k + 32, 0, 3);
      __builtin_prefetch(Bthi + boff[0] + k + 32, 0, 3);
    }
    v16bf ah[2], al[2], bh[4], bl[4];
#pragma unroll
    for (int i = 0; i < 2; ++i) {
      ah[i] = fragA(Ahi + aoff[i], k, half);
      al[i] = fragA(Alo + aoff[i], k, half);
    }
#pragma unroll
    for (int j = 0; j < 4; ++j) {
      bh[j] = fragB(Bthi + boff[j], k, half);
      bl[j] = fragB(Btlo + boff[j], k, half);
    }
#pragma unroll
    for (int i = 0; i < 2; ++i)
#pragma unroll
      for (int j = 0; j < 4; ++j) {
        acc[i][j] = __builtin_amdgcn_wmma_f32_16x16x32_bf16(
            false, ah[i], false, bh[j], (short)0, acc[i][j], false, false);
        acc[i][j] = __builtin_amdgcn_wmma_f32_16x16x32_bf16(
            false, ah[i], false, bl[j], (short)0, acc[i][j], false, false);
        acc[i][j] = __builtin_amdgcn_wmma_f32_16x16x32_bf16(
            false, al[i], false, bh[j], (short)0, acc[i][j], false, false);
      }
  }

  // C/D layout: VGPR v -> M = v + 8*half ; lane r -> N
  if (Chi) {
#pragma unroll
    for (int i = 0; i < 2; ++i)
#pragma unroll
      for (int j = 0; j < 4; ++j) {
        int n = col0 + 16 * j + r;
        float bv = bias[n];
#pragma unroll
        for (int v = 0; v < 8; ++v) {
          int m = row0 + 16 * i + v + 8 * half;
          float val = fmaxf(acc[i][j][v] + bv, 0.f);
          long o = (long)m * Nc + n;
          __bf16 hb = (__bf16)val;
          Chi[o] = hb;
          Clo[o] = (__bf16)(val - (float)hb);
        }
      }
  }
  if (Pout) {
    // fused out-projection: s(m) = sum_n relu(acc+bias)*Wv[n] over this wave's 64 cols.
    // Reduce over lanes r=0..15 inside each half (xor<16 never crosses the half).
    const int nslots = gridDim.y * 2;
    const int slot   = blockIdx.y * 2 + wn;
#pragma unroll
    for (int i = 0; i < 2; ++i)
#pragma unroll
      for (int v = 0; v < 8; ++v) {
        float s = 0.f;
#pragma unroll
        for (int j = 0; j < 4; ++j) {
          int n = col0 + 16 * j + r;
          s += fmaxf(acc[i][j][v] + bias[n], 0.f) * Wv[n];
        }
#pragma unroll
        for (int off = 1; off < 16; off <<= 1) s += __shfl_xor(s, off, 32);
        if (r == 0) {
          int m = row0 + 16 * i + v + 8 * half;
          Pout[(long)m * nslots + slot] = s;
        }
      }
  }
}

// ---------------- kernel 6: out = sum(partials) + xpart + bout (fixed order) ----------
__global__ void final_sum_kernel(const float* __restrict__ partial,
                                 const float* __restrict__ xpart,
                                 const float* __restrict__ bout,
                                 float* __restrict__ out, int nslots) {
  int row = blockIdx.x * blockDim.x + threadIdx.x;
  if (row >= Bz) return;
  float s = xpart[row] + bout[0];
  const float* p = partial + (long)row * nslots;
  for (int i = 0; i < nslots; ++i) s += p[i];
  out[row] = s;
}

// ---------------- host launcher ----------------
extern "C" void kernel_launch(void* const* d_in, const int* in_sizes, int n_in,
                              void* d_out, int out_size, void* d_ws, size_t ws_size,
                              hipStream_t stream) {
  const int*   Xi     = (const int*)d_in[0];     // [B,F,1] int32
  const float* Xv     = (const float*)d_in[1];   // [B,13]
  const float* emb    = (const float*)d_in[2];   // [F*N, 16]
  const float* crossw = (const float*)d_in[3];   // [L, D]
  const float* crossb = (const float*)d_in[4];   // [L, D]
  const float* W1     = (const float*)d_in[5];   // [D, H]
  const float* b1     = (const float*)d_in[6];   // [H]
  const float* W2     = (const float*)d_in[7];   // [H, H]
  const float* b2     = (const float*)d_in[8];   // [H]
  const float* Wout   = (const float*)d_in[9];   // [H+D, 1]
  const float* bout   = (const float*)d_in[10];  // [1]
  float* out = (float*)d_out;                    // [B]

  // workspace partition (256B aligned slabs), ~135 MB total
  char* ws = (char*)d_ws;
  size_t off = 0;
  auto alloc = [&](size_t bytes) -> void* {
    void* p = ws + off;
    off = (off + bytes + 255) & ~(size_t)255;
    return p;
  };
  constexpr int PSLOTS = (Hz / 128) * 2;  // 16 exclusive partial slots per row
  float*  x0f   = (float*)alloc((size_t)Bz * DPz * 4);
  __bf16* x0hi  = (__bf16*)alloc((size_t)Bz * DPz * 2);
  __bf16* x0lo  = (__bf16*)alloc((size_t)Bz * DPz * 2);
  float*  xpart = (float*)alloc((size_t)Bz * 4);
  __bf16* w1thi = (__bf16*)alloc((size_t)Hz * DPz * 2);
  __bf16* w1tlo = (__bf16*)alloc((size_t)Hz * DPz * 2);
  __bf16* w2thi = (__bf16*)alloc((size_t)Hz * Hz * 2);
  __bf16* w2tlo = (__bf16*)alloc((size_t)Hz * Hz * 2);
  __bf16* h1hi  = (__bf16*)alloc((size_t)Bz * Hz * 2);
  __bf16* h1lo  = (__bf16*)alloc((size_t)Bz * Hz * 2);
  float*  part  = (float*)alloc((size_t)Bz * PSLOTS * 4);
  (void)ws_size; (void)in_sizes; (void)n_in; (void)out_size;

  // 1) x0 = concat(Xv, embedding gather), padded + bf16 split
  build_x0_kernel<<<(Bz * DPz + 255) / 256, 256, 0, stream>>>(Xi, Xv, emb, x0f, x0hi, x0lo);
  // 2) weight transpose/pad/split (cheap, per-call; deterministic)
  cvt_wT_kernel<<<(Hz * DPz + 255) / 256, 256, 0, stream>>>(W1, w1thi, w1tlo, Hz, DPz, Dz);
  cvt_wT_kernel<<<(Hz * Hz + 255) / 256, 256, 0, stream>>>(W2, w2thi, w2tlo, Hz, Hz, Hz);
  // 3) CrossNet -> per-row scalar contribution to the output
  crossnet_kernel<<<Bz / 8, 256, 0, stream>>>(x0f, crossw, crossb, Wout, xpart);
  // 4) h1 = relu(x0 @ W1 + b1)  [16384x448x1024] -> bf16 hi/lo
  gemm_bf16x3_kernel<<<dim3(Bz / 128, Hz / 128), 256, 0, stream>>>(
      x0hi, x0lo, w1thi, w1tlo, b1, h1hi, h1lo, nullptr, nullptr, DPz, Hz);
  // 5) h2 = relu(h1 @ W2 + b2) fused with h2.Wout[:H] -> deterministic partial slots
  gemm_bf16x3_kernel<<<dim3(Bz / 128, Hz / 128), 256, 0, stream>>>(
      h1hi, h1lo, w2thi, w2tlo, b2, nullptr, nullptr, part, Wout, Hz, Hz);
  // 6) out = sum(partials) + xpart + bout
  final_sum_kernel<<<(Bz + 255) / 256, 256, 0, stream>>>(part, xpart, bout, out, PSLOTS);
}
